// LSTMModule_39067022524827
// MI455X (gfx1250) — matrix operand order repaired and checked by reading, not compiled, not run on hardware
//
#include <hip/hip_runtime.h>

#define B_   128
#define T_   512
#define F_   512
#define H1_  1024
#define H2_  512
#define KTOT 1536
#define NC1  (4 * H1_)   // 4096
#define NC2  (4 * H2_)   // 2048

typedef __attribute__((ext_vector_type(16))) __bf16 v16bf;
typedef __attribute__((ext_vector_type(8)))  float  v8f;

// ---- workspace layout (bytes) ----
#define O_W1T  ((size_t)0)           // 4096*1536*2 = 12,582,912
#define O_W2T  ((size_t)12582912)    // 2048*1536*2 =  6,291,456
#define O_B1R  ((size_t)18874368)    // 4096*4
#define O_B2R  ((size_t)18890752)    // 2048*4
#define O_ST   ((size_t)18898944)
#define O_H1   (O_ST + 0)            // 2*128*1024 bf16 = 524,288 B (ping-pong)
#define O_C1   (O_ST + 524288)       // 128*1024 f32    = 524,288 B
#define O_H2   (O_ST + 1048576)      // 2*128*512 bf16  = 262,144 B (ping-pong)
#define O_C2   (O_ST + 1310720)      // 128*512 f32
#define O_H2F  (O_ST + 1572864)      // 128*512 f32 (last-step h2 for projection)
#define ST_BYTES 1835008

// ---------------- WMMA helpers ----------------
__device__ __forceinline__ v8f wmma_bf16(v16bf a, v16bf b, v8f c) {
  return __builtin_amdgcn_wmma_f32_16x16x32_bf16(false, a, false, b, (short)0, c,
                                                 false, false);
}

// A fragment (16x32 bf16): lane holds row m = lmod; element j -> K = (j<8 ?
// lhalf*8 + j : 16 + lhalf*8 + j-8). Two contiguous 8-elem (16B) runs.
__device__ __forceinline__ v16bf load_a_bf16(const __bf16* row, int kb, int lhalf) {
  union { v16bf v; uint4 q[2]; } u;
  u.q[0] = *(const uint4*)(row + kb + lhalf * 8);
  u.q[1] = *(const uint4*)(row + kb + 16 + lhalf * 8);
  return u.v;
}

__device__ __forceinline__ v16bf load_a_f32(const float* row, int kb, int lhalf) {
  float4 f0 = *(const float4*)(row + kb + lhalf * 8);
  float4 f1 = *(const float4*)(row + kb + lhalf * 8 + 4);
  float4 f2 = *(const float4*)(row + kb + 16 + lhalf * 8);
  float4 f3 = *(const float4*)(row + kb + 16 + lhalf * 8 + 4);
  v16bf a;
  a[0] = (__bf16)f0.x;  a[1] = (__bf16)f0.y;  a[2] = (__bf16)f0.z;  a[3] = (__bf16)f0.w;
  a[4] = (__bf16)f1.x;  a[5] = (__bf16)f1.y;  a[6] = (__bf16)f1.z;  a[7] = (__bf16)f1.w;
  a[8] = (__bf16)f2.x;  a[9] = (__bf16)f2.y;  a[10] = (__bf16)f2.z; a[11] = (__bf16)f2.w;
  a[12] = (__bf16)f3.x; a[13] = (__bf16)f3.y; a[14] = (__bf16)f3.z; a[15] = (__bf16)f3.w;
  return a;
}

// B fragment (32x16 bf16): lane holds col n = lmod; element j -> K = (j<8 ?
// lhalf*8 + j : 16 + lhalf*8 + j-8). Weights stored transposed [NC][KTOT]: two 16B runs.
__device__ __forceinline__ v16bf load_b(const __bf16* Wt, int nc, int k, int lhalf) {
  const __bf16* p = Wt + (size_t)nc * KTOT + k + lhalf * 8;
  union { v16bf v; uint4 q[2]; } u;
  u.q[0] = *(const uint4*)p;
  u.q[1] = *(const uint4*)(p + 16);
  return u.v;
}

// Fused LSTM cell update. acc[i][g]: i = M-fragment (rows m0+i*16..), g = gate
// (i,f,g,o) thanks to the gate-interleaved weight layout. C/D frag layout:
// lane L, vgpr r -> row m0 + i*16 + (L>>4)*8 + r, col = hidden hb*16 + (L&15).
__device__ __forceinline__ void cell_update(v8f (&acc)[2][4],
                                            const float* __restrict__ br,
                                            float* __restrict__ cst,
                                            __bf16* __restrict__ hout,
                                            float* __restrict__ hfout,
                                            int m0, int hb, int lhalf, int lmod, int H) {
  float bg[4];
#pragma unroll
  for (int g = 0; g < 4; ++g) bg[g] = br[hb * 64 + g * 16 + lmod];
  const int u = hb * 16 + lmod;
#pragma unroll
  for (int i = 0; i < 2; ++i) {
#pragma unroll
    for (int r = 0; r < 8; ++r) {
      int m = m0 + i * 16 + lhalf * 8 + r;
      size_t idx = (size_t)m * H + u;
      float iv = 1.f / (1.f + __expf(-(acc[i][0][r] + bg[0])));
      float fv = 1.f / (1.f + __expf(-(acc[i][1][r] + bg[1])));
      float gv = tanhf(acc[i][2][r] + bg[2]);
      float ov = 1.f / (1.f + __expf(-(acc[i][3][r] + bg[3])));
      float cn = fv * cst[idx] + iv * gv;
      cst[idx] = cn;
      float hv = ov * tanhf(cn);
      hout[idx] = (__bf16)hv;
      if (hfout) hfout[idx] = hv;
    }
  }
}

// One pipelined time step: blocks 0..31 do layer-1 step t; blocks 32..47 do
// layer-2 step t-1 (consuming h1 written by the previous launch; ping-pong
// buffers make the cross-layer overlap race-free).
__global__ __launch_bounds__(256) void lstm_step_kernel(
    const float* __restrict__ x,
    const __bf16* __restrict__ W1t, const float* __restrict__ b1r,
    const __bf16* __restrict__ W2t, const float* __restrict__ b2r,
    __bf16* __restrict__ h1buf, float* __restrict__ c1,
    __bf16* __restrict__ h2buf, float* __restrict__ c2,
    float* __restrict__ h2f, int t) {
  const int wg = blockIdx.x;
  const int wave = threadIdx.x >> 5;
  const int lane = threadIdx.x & 31;
  const int lhalf = lane >> 4;
  const int lmod = lane & 15;

  v8f acc[2][4] = {};

  if (wg < 32) {
    // ---- layer 1, step t: gates = [x_t, h1_{t-1}] @ W1 ----
    if (t >= T_) return;
    const int mblk = wg & 3;
    const int hb = ((wg >> 2) << 3) + wave;   // 0..63 hidden block
    const int m0 = mblk * 32;
    const float* x0 = x + (size_t)(m0 + lmod) * (T_ * F_) + (size_t)t * F_;
    const float* x1 = x0 + (size_t)16 * (T_ * F_);
    const __bf16* hp = h1buf + (size_t)(1 - (t & 1)) * B_ * H1_;
    const __bf16* h0 = hp + (size_t)(m0 + lmod) * H1_;
    const __bf16* h1r = h0 + (size_t)16 * H1_;
    const int colbase = hb * 64 + lmod;

    for (int kb = 0; kb < F_; kb += 32) {          // x part (f32 -> bf16)
      v16bf a0 = load_a_f32(x0, kb, lhalf);
      v16bf a1 = load_a_f32(x1, kb, lhalf);
#pragma unroll
      for (int g = 0; g < 4; ++g) {
        v16bf bf = load_b(W1t, colbase + g * 16, kb, lhalf);
        acc[0][g] = wmma_bf16(a0, bf, acc[0][g]);
        acc[1][g] = wmma_bf16(a1, bf, acc[1][g]);
      }
    }
    for (int kb = 0; kb < H1_; kb += 32) {         // recurrent part
      v16bf a0 = load_a_bf16(h0, kb, lhalf);
      v16bf a1 = load_a_bf16(h1r, kb, lhalf);
#pragma unroll
      for (int g = 0; g < 4; ++g) {
        v16bf bf = load_b(W1t, colbase + g * 16, F_ + kb, lhalf);
        acc[0][g] = wmma_bf16(a0, bf, acc[0][g]);
        acc[1][g] = wmma_bf16(a1, bf, acc[1][g]);
      }
    }
    __bf16* hout = h1buf + (size_t)(t & 1) * B_ * H1_;
    cell_update(acc, b1r, c1, hout, nullptr, m0, hb, lhalf, lmod, H1_);
  } else {
    // ---- layer 2, step s = t-1: gates = [h1_s, h2_{s-1}] @ W2 ----
    if (t < 1) return;
    const int s = t - 1;
    const int wg2 = wg - 32;
    const int mblk = wg2 & 3;
    const int hb = ((wg2 >> 2) << 3) + wave;  // 0..31
    const int m0 = mblk * 32;
    const __bf16* ys = h1buf + (size_t)(s & 1) * B_ * H1_;
    const __bf16* y0 = ys + (size_t)(m0 + lmod) * H1_;
    const __bf16* y1 = y0 + (size_t)16 * H1_;
    const __bf16* hp = h2buf + (size_t)(1 - (s & 1)) * B_ * H2_;
    const __bf16* h0 = hp + (size_t)(m0 + lmod) * H2_;
    const __bf16* h1r = h0 + (size_t)16 * H2_;
    const int colbase = hb * 64 + lmod;

    for (int kb = 0; kb < H1_; kb += 32) {
      v16bf a0 = load_a_bf16(y0, kb, lhalf);
      v16bf a1 = load_a_bf16(y1, kb, lhalf);
#pragma unroll
      for (int g = 0; g < 4; ++g) {
        v16bf bf = load_b(W2t, colbase + g * 16, kb, lhalf);
        acc[0][g] = wmma_bf16(a0, bf, acc[0][g]);
        acc[1][g] = wmma_bf16(a1, bf, acc[1][g]);
      }
    }
    for (int kb = 0; kb < H2_; kb += 32) {
      v16bf a0 = load_a_bf16(h0, kb, lhalf);
      v16bf a1 = load_a_bf16(h1r, kb, lhalf);
#pragma unroll
      for (int g = 0; g < 4; ++g) {
        v16bf bf = load_b(W2t, colbase + g * 16, H1_ + kb, lhalf);
        acc[0][g] = wmma_bf16(a0, bf, acc[0][g]);
        acc[1][g] = wmma_bf16(a1, bf, acc[1][g]);
      }
    }
    __bf16* hout = h2buf + (size_t)(s & 1) * B_ * H2_;
    cell_update(acc, b2r, c2, hout, h2f, m0, hb, lhalf, lmod, H2_);
  }
}

// W[k][gate*H + u] (f32, [KTOT, 4H]) -> Wt[nc][k] (bf16, [4H, KTOT]) with
// gate-interleaved columns: nc = (u/16)*64 + gate*16 + (u%16).
__global__ void convert_w_kernel(const float* __restrict__ W, __bf16* __restrict__ Wt,
                                 int H, int NC) {
  size_t idx = (size_t)blockIdx.x * blockDim.x + threadIdx.x;
  if (idx >= (size_t)NC * KTOT) return;
  int nc = (int)(idx / KTOT);
  int k = (int)(idx % KTOT);
  int hbl = nc >> 6, g = (nc >> 4) & 3, j = nc & 15;
  int u = hbl * 16 + j;
  Wt[idx] = (__bf16)W[(size_t)k * NC + (g * H + u)];
}

__global__ void convert_b_kernel(const float* __restrict__ b, float* __restrict__ br,
                                 int H, int NC) {
  int nc = blockIdx.x * blockDim.x + threadIdx.x;
  if (nc >= NC) return;
  int hbl = nc >> 6, g = (nc >> 4) & 3, j = nc & 15;
  br[nc] = b[g * H + hbl * 16 + j];
}

__global__ void zero_kernel(uint4* __restrict__ p, int n16) {
  int i = blockIdx.x * blockDim.x + threadIdx.x;
  if (i < n16) p[i] = make_uint4(0u, 0u, 0u, 0u);
}

__global__ void final_proj_kernel(const float* __restrict__ h2f,
                                  const float* __restrict__ Wout,
                                  const float* __restrict__ bout,
                                  float* __restrict__ out) {
  __shared__ float red[256];
  int b = blockIdx.x, tid = threadIdx.x;
  float s = h2f[(size_t)b * H2_ + tid] * Wout[tid] +
            h2f[(size_t)b * H2_ + 256 + tid] * Wout[256 + tid];
  red[tid] = s;
  __syncthreads();
  for (int off = 128; off > 0; off >>= 1) {
    if (tid < off) red[tid] += red[tid + off];
    __syncthreads();
  }
  if (tid == 0) out[b] = red[0] + bout[0];
}

extern "C" void kernel_launch(void* const* d_in, const int* in_sizes, int n_in,
                              void* d_out, int out_size, void* d_ws, size_t ws_size,
                              hipStream_t stream) {
  (void)in_sizes; (void)n_in; (void)out_size; (void)ws_size;
  const float* x    = (const float*)d_in[0];
  const float* W1   = (const float*)d_in[1];
  const float* b1   = (const float*)d_in[2];
  const float* W2   = (const float*)d_in[3];
  const float* b2   = (const float*)d_in[4];
  const float* Wout = (const float*)d_in[5];
  const float* bout = (const float*)d_in[6];
  float* out = (float*)d_out;
  char* ws = (char*)d_ws;

  __bf16* W1t  = (__bf16*)(ws + O_W1T);
  __bf16* W2t  = (__bf16*)(ws + O_W2T);
  float* b1r   = (float*)(ws + O_B1R);
  float* b2r   = (float*)(ws + O_B2R);
  __bf16* h1b  = (__bf16*)(ws + O_H1);
  float* c1    = (float*)(ws + O_C1);
  __bf16* h2b  = (__bf16*)(ws + O_H2);
  float* c2    = (float*)(ws + O_C2);
  float* h2f   = (float*)(ws + O_H2F);

  zero_kernel<<<ST_BYTES / 16 / 256, 256, 0, stream>>>((uint4*)(ws + O_ST),
                                                       ST_BYTES / 16);
  convert_w_kernel<<<((size_t)NC1 * KTOT + 255) / 256, 256, 0, stream>>>(W1, W1t, H1_, NC1);
  convert_w_kernel<<<((size_t)NC2 * KTOT + 255) / 256, 256, 0, stream>>>(W2, W2t, H2_, NC2);
  convert_b_kernel<<<(NC1 + 255) / 256, 256, 0, stream>>>(b1, b1r, H1_, NC1);
  convert_b_kernel<<<(NC2 + 255) / 256, 256, 0, stream>>>(b2, b2r, H2_, NC2);

  // T+1 pipelined launches: launch t runs layer-1 step t and layer-2 step t-1.
  for (int t = 0; t <= T_; ++t)
    lstm_step_kernel<<<48, 256, 0, stream>>>(x, W1t, b1r, W2t, b2r, h1b, c1, h2b,
                                             c2, h2f, t);

  final_proj_kernel<<<B_, 256, 0, stream>>>(h2f, Wout, bout, out);
}
